// EfficientAttention_45114336477532
// MI455X (gfx1250) — matrix operand order repaired and checked by reading, not compiled, not run on hardware
//
#include <hip/hip_runtime.h>
#include <hip/hip_bf16.h>

typedef __bf16 bf16_t;
typedef __attribute__((ext_vector_type(16))) __bf16 v16bf;
typedef __attribute__((ext_vector_type(8)))  float  v8f;

// Exact RNE conversion (used for one-time weight staging)
static __device__ __forceinline__ bf16_t f2bf(float f) {
    unsigned u = __float_as_uint(f);
    u += 0x7FFFu + ((u >> 16) & 1u);
    unsigned short h = (unsigned short)(u >> 16);
    bf16_t r;
    __builtin_memcpy(&r, &h, sizeof(r));
    return r;
}

// Fast packed conversion: two f32 -> one dword holding {bf16(hi), bf16(lo)}.
// 2x v_add + 1x v_perm_b32 instead of ~6 VALU for two scalar conversions.
static __device__ __forceinline__ unsigned f2bf2(float lo, float hi) {
    unsigned ul = __float_as_uint(lo) + 0x8000u;   // round-half-up
    unsigned uh = __float_as_uint(hi) + 0x8000u;
    return __builtin_amdgcn_perm(uh, ul, 0x07060302u); // {uh[31:16], ul[31:16]}
}

// ---------------------------------------------------------------------------
// Tensor Data Mover: DMA a [rows x cols] f32 tile (row stride = stride elts)
// from global memory into LDS at lds_byte_addr (contiguous row-major).
// D# layout per CDNA5 ISA.
// ---------------------------------------------------------------------------
#if __has_builtin(__builtin_amdgcn_tensor_load_to_lds)
#define HAS_TDM 1
typedef unsigned int tdm_v4u __attribute__((ext_vector_type(4)));
typedef int          tdm_v8i __attribute__((ext_vector_type(8)));
typedef int          tdm_v4i __attribute__((ext_vector_type(4)));

static __device__ __forceinline__ void tdm_load_tile_f32(
    const float* gsrc, unsigned lds_byte_addr, int rows, int cols, int stride)
{
    unsigned long long ga = (unsigned long long)(size_t)gsrc;
    // group0: count=1 | lds_addr | global_addr[56:0] | type=2
    tdm_v4u g0 = { 1u,
                   lds_byte_addr,
                   (unsigned)ga,
                   (((unsigned)(ga >> 32)) & 0x01FFFFFFu) | 0x80000000u };
    // group1: wg_mask=0, data_size=2(4B); tensor_dim0/1 = 1<<20 (never OOB);
    //         tile_dim0=cols, tile_dim1=rows, tile_dim2=0; dim0_stride=stride
    const unsigned td = 1u << 20;
    tdm_v8i g1 = { (int)(2u << 16),
                   (int)((td & 0xFFFFu) << 16),
                   (int)((td >> 16) | ((td & 0xFFFFu) << 16)),
                   (int)((td >> 16) | ((unsigned)cols << 16)),
                   rows,
                   stride,
                   0, 0 };
    tdm_v4i gz = { 0, 0, 0, 0 };
#if __has_include(<hip/amd_detail/amd_gfx1250_TDM.h>)
    tdm_v8i gz8 = { 0, 0, 0, 0, 0, 0, 0, 0 };
    __builtin_amdgcn_tensor_load_to_lds(g0, g1, gz, gz, gz8, 0);
#else
    __builtin_amdgcn_tensor_load_to_lds(g0, g1, gz, gz, 0);
#endif
}
#else
#define HAS_TDM 0
#endif

// ---------------------------------------------------------------------------
// Pointwise conv as batched GEMM:  Y[n][m][l] = sum_k W[m][k] * X[n][k][l] + B[m]
// Block = 128 threads (4 waves) -> one 16(M) x 64(L) macro-tile.
// ---------------------------------------------------------------------------
__global__ void pw_gemm_wmma(const float* __restrict__ X, const float* __restrict__ Wt,
                             const float* __restrict__ Bias, float* __restrict__ Y,
                             int M, int K, int Lw, int mtiles, int l4tiles)
{
    __shared__ __align__(16) unsigned short a_lds[16 * 258]; // stride K+2: no bank conflicts
    __shared__ __align__(16) float          b_lds[32 * 64];

    int bid = blockIdx.x;
    int lt4 = bid % l4tiles;
    int t2  = bid / l4tiles;
    int mt  = t2 % mtiles;
    int n   = t2 / mtiles;

    int tid  = threadIdx.x;
    int wv   = tid >> 5;
    int lane = tid & 31;
    int half = lane >> 4;
    int l16  = lane & 15;
    int astr = K + 2;

    const float* Xn = X + (size_t)n * K * Lw;
    float*       Yn = Y + (size_t)n * M * Lw;
    int lcol0 = lt4 * 64;
    int c     = (wv << 4) + l16;          // this lane's column within the 64-wide slab

    // --- stage weight tile [16 x K] as bf16 into LDS (coalesced over k, RNE) ---
    for (int i = tid; i < 16 * K; i += 128) {
        int m = i / K, k = i % K;
        bf16_t bv = f2bf(Wt[(size_t)(mt * 16 + m) * K + k]);
        unsigned short hb;
        __builtin_memcpy(&hb, &bv, 2);
        a_lds[m * astr + k] = hb;
    }
    __syncthreads();

#if HAS_TDM
    unsigned b_lds_addr = (unsigned)(size_t)&b_lds[0];
#endif

    v8f acc = {};
    for (int k0 = 0; k0 < K; k0 += 32) {
        if (k0) __syncthreads();                 // previous tile fully consumed
        // --- stage X tile [32 k x 64 l] into LDS ---
#if HAS_TDM
        if (tid < 32) {
            tdm_load_tile_f32(Xn + (size_t)k0 * Lw + lcol0, b_lds_addr, 32, 64, Lw);
            __builtin_amdgcn_s_wait_tensorcnt(0);
        }
#else
        if (k0 + 32 < K)
            __builtin_prefetch(Xn + (size_t)(k0 + 32) * Lw + lcol0 + (tid & 63), 0, 1);
        for (int i = tid; i < 32 * 64; i += 128) {
            int r = i >> 6, cc = i & 63;
            b_lds[i] = Xn[(size_t)(k0 + r) * Lw + lcol0 + cc];
        }
#endif
        __syncthreads();

        // --- A fragment: packed bf16 pairs straight from LDS ---
        union { unsigned u[8]; v16bf v; } ua;
#pragma unroll
        for (int p = 0; p < 8; ++p) {
            int ka = k0 + 2 * p + 8 * half + ((p >> 2) << 3);
            ua.u[p] = *(const unsigned*)&a_lds[l16 * astr + ka];
        }
        // --- B fragment: f32 pairs from LDS, packed conversion ---
        union { unsigned u[8]; v16bf v; } ub;
#pragma unroll
        for (int p = 0; p < 8; ++p) {
            int kb = 2 * p + 8 * half + ((p >> 2) << 3);
            ub.u[p] = f2bf2(b_lds[kb * 64 + c], b_lds[(kb + 1) * 64 + c]);
        }

        acc = __builtin_amdgcn_wmma_f32_16x16x32_bf16(false, ua.v, false, ub.v,
                                                      (short)0, acc, false, false);
    }

    int col = lcol0 + c;
#pragma unroll
    for (int r = 0; r < 8; ++r) {
        int m = (mt << 4) + r + 8 * half;
        Yn[(size_t)m * Lw + col] = acc[r] + Bias[m];
    }
}

// ---------------------------------------------------------------------------
// Depthwise 3x3x3 conv, pad=1, channels=32, shape [N,32,16,28,28]
// ---------------------------------------------------------------------------
__global__ void dw_conv3(const float* __restrict__ x, const float* __restrict__ w,
                         const float* __restrict__ bias, float* __restrict__ y, int total)
{
    int idx = blockIdx.x * blockDim.x + threadIdx.x;
    if (idx >= total) return;
    int iw = idx % 28; int tmp = idx / 28;
    int ih = tmp % 28; tmp /= 28;
    int it = tmp % 16; tmp /= 16;
    int m  = tmp % 32; int nn = tmp / 32;
    const float* xb = x + (size_t)(nn * 32 + m) * (16 * 28 * 28);
    const float* wm = w + m * 27;
    float acc = bias[m];
#pragma unroll
    for (int dt = -1; dt <= 1; ++dt)
#pragma unroll
        for (int dh = -1; dh <= 1; ++dh)
#pragma unroll
            for (int dw = -1; dw <= 1; ++dw) {
                int t2 = it + dt, h2 = ih + dh, w2 = iw + dw;
                if ((unsigned)t2 < 16u && (unsigned)h2 < 28u && (unsigned)w2 < 28u)
                    acc += wm[(dt + 1) * 9 + (dh + 1) * 3 + (dw + 1)] *
                           xb[(t2 * 28 + h2) * 28 + w2];
            }
    y[idx] = acc;
}

// ---------------------------------------------------------------------------
// Softmax over L per row (K path): 512 rows, one block per row
// ---------------------------------------------------------------------------
__global__ void softmax_rows(float* __restrict__ data, int Lw)
{
    __shared__ float red[256];
    float* p = data + (size_t)blockIdx.x * Lw;
    int tid = threadIdx.x;
    float m = -1e30f;
    for (int i = tid; i < Lw; i += 256) m = fmaxf(m, p[i]);
    red[tid] = m; __syncthreads();
    for (int s = 128; s > 0; s >>= 1) { if (tid < s) red[tid] = fmaxf(red[tid], red[tid + s]); __syncthreads(); }
    m = red[0]; __syncthreads();
    float sum = 0.f;
    for (int i = tid; i < Lw; i += 256) sum += __expf(p[i] - m);
    red[tid] = sum; __syncthreads();
    for (int s = 128; s > 0; s >>= 1) { if (tid < s) red[tid] += red[tid + s]; __syncthreads(); }
    float inv = 1.f / red[0];
    for (int i = tid; i < Lw; i += 256) p[i] = __expf(p[i] - m) * inv;
}

// ---------------------------------------------------------------------------
// Softmax over the 16 per-head key channels (Q path), one thread per (n,h,l)
// ---------------------------------------------------------------------------
__global__ void softmax_chan(float* __restrict__ Q, int total, int Lw)
{
    int idx = blockIdx.x * blockDim.x + threadIdx.x;
    if (idx >= total) return;
    int l = idx % Lw; int h = (idx / Lw) & 7; int n = idx / (Lw * 8);
    float* p = Q + (size_t)(n * 128 + h * 16) * Lw + l;
    float m = -1e30f;
    for (int k = 0; k < 16; ++k) m = fmaxf(m, p[(size_t)k * Lw]);
    float s = 0.f;
    for (int k = 0; k < 16; ++k) s += __expf(p[(size_t)k * Lw] - m);
    float inv = 1.f / s;
    for (int k = 0; k < 16; ++k) p[(size_t)k * Lw] = __expf(p[(size_t)k * Lw] - m) * inv;
}

// ---------------------------------------------------------------------------
// context[n,h,kk,vv] = sum_l K[n,h,kk,l] * V[n,h,vv,l]   (16x32 per head)
// Fragment pairs are contiguous in memory -> float2 loads + packed conversion.
// ---------------------------------------------------------------------------
__global__ void ctx_wmma(const float* __restrict__ Kb, const float* __restrict__ Vb,
                         float* __restrict__ ctx, int Lw)
{
    int tile = blockIdx.x * 4 + (threadIdx.x >> 5);
    if (tile >= 64) return;
    int lane = threadIdx.x & 31;
    int vt = tile & 1; int h = (tile >> 1) & 7; int n = tile >> 4;
    int half = lane >> 4, l16 = lane & 15;
    const float* Kr = Kb + (size_t)(n * 128 + h * 16 + l16) * Lw;
    const float* Vr = Vb + (size_t)(n * 256 + h * 32 + vt * 16 + l16) * Lw;
    v8f acc = {};
    for (int l0 = 0; l0 < Lw; l0 += 32) {
        union { unsigned u[8]; v16bf v; } ua, ub;
#pragma unroll
        for (int p = 0; p < 8; ++p) {
            float2 av = *(const float2*)&Kr[l0 + 2 * p + 8 * half + ((p >> 2) << 3)];
            ua.u[p] = f2bf2(av.x, av.y);
            float2 bv = *(const float2*)&Vr[l0 + 2 * p + 8 * half + ((p >> 2) << 3)];
            ub.u[p] = f2bf2(bv.x, bv.y);
        }
        acc = __builtin_amdgcn_wmma_f32_16x16x32_bf16(false, ua.v, false, ub.v,
                                                      (short)0, acc, false, false);
    }
#pragma unroll
    for (int r = 0; r < 8; ++r) {
        int kk = r + 8 * half;
        ctx[((size_t)(n * 8 + h) * 16 + kk) * 32 + vt * 16 + l16] = acc[r];
    }
}

// ---------------------------------------------------------------------------
// attended[n,h,vv,l] = sum_kk ctx[n,h,kk,vv] * Q[n,h,kk,l]  (K=16 zero-padded)
// ---------------------------------------------------------------------------
__global__ void att_wmma(const float* __restrict__ ctx, const float* __restrict__ Qb,
                         float* __restrict__ agg, int Lw, int totalTiles)
{
    int tile = blockIdx.x * 4 + (threadIdx.x >> 5);
    if (tile >= totalTiles) return;
    int lane = threadIdx.x & 31;
    int ltiles = Lw >> 4;
    int lt = tile % ltiles; int t2 = tile / ltiles;
    int vt = t2 & 1; int h = (t2 >> 1) & 7; int n = t2 >> 4;
    int half = lane >> 4, l16 = lane & 15;
    int vv = vt * 16 + l16;
    const float* cb = ctx + (size_t)(n * 8 + h) * 16 * 32;
    const float* qb = Qb + (size_t)(n * 128 + h * 16) * Lw + lt * 16 + l16;
    union { unsigned u[8]; v16bf v; } ua, ub;
#pragma unroll
    for (int p = 0; p < 8; ++p) {
        int ka = 2 * p + 8 * half + ((p >> 2) << 3);      // valid (<16) iff p < 4
        ua.u[p] = (p < 4) ? f2bf2(cb[ka * 32 + vv], cb[(ka + 1) * 32 + vv]) : 0u;
        ub.u[p] = (p < 4) ? f2bf2(qb[(size_t)(ka) * Lw],
                                  qb[(size_t)(ka + 1) * Lw]) : 0u;
    }
    v8f acc = {};
    acc = __builtin_amdgcn_wmma_f32_16x16x32_bf16(false, ua.v, false, ub.v,
                                                  (short)0, acc, false, false);
#pragma unroll
    for (int r = 0; r < 8; ++r) {
        int m = vt * 16 + r + 8 * half;
        agg[(size_t)(n * 256 + h * 32 + m) * Lw + lt * 16 + l16] = acc[r];
    }
}

// ---------------------------------------------------------------------------
// x = reproj + input ; deterministic two-level reduction for global layernorm
// ---------------------------------------------------------------------------
__global__ void add_reduce(const float* __restrict__ rp, const float* __restrict__ in,
                           float* __restrict__ xb, float* __restrict__ part, int total)
{
    __shared__ float rs[256], rs2[256];
    int tid = threadIdx.x;
    size_t stride = (size_t)gridDim.x * blockDim.x;
    float s = 0.f, s2 = 0.f;
    for (size_t i = (size_t)blockIdx.x * blockDim.x + tid; i < (size_t)total; i += stride) {
        float v = rp[i] + in[i];
        xb[i] = v;
        s += v; s2 += v * v;
    }
    rs[tid] = s; rs2[tid] = s2; __syncthreads();
    for (int st = 128; st > 0; st >>= 1) {
        if (tid < st) { rs[tid] += rs[tid + st]; rs2[tid] += rs2[tid + st]; }
        __syncthreads();
    }
    if (tid == 0) { part[blockIdx.x] = rs[0]; part[1024 + blockIdx.x] = rs2[0]; }
}

__global__ void finalize_stats(float* __restrict__ part, int total)
{
    __shared__ float rs[256], rs2[256];
    int tid = threadIdx.x;
    float s = 0.f, s2 = 0.f;
    for (int i = tid; i < 1024; i += 256) { s += part[i]; s2 += part[1024 + i]; }
    rs[tid] = s; rs2[tid] = s2; __syncthreads();
    for (int st = 128; st > 0; st >>= 1) {
        if (tid < st) { rs[tid] += rs[tid + st]; rs2[tid] += rs2[tid + st]; }
        __syncthreads();
    }
    if (tid == 0) {
        float mu  = rs[0] / (float)total;
        float var = rs2[0] / (float)total - mu * mu;
        part[2048] = mu;
        part[2049] = rsqrtf(var + 1e-5f);
    }
}

__global__ void normalize_out(const float* __restrict__ xb, const float* __restrict__ part,
                              float* __restrict__ out, int total)
{
    size_t i = (size_t)blockIdx.x * blockDim.x + threadIdx.x;
    if (i >= (size_t)total) return;
    float mu = part[2048], rstd = part[2049];
    out[i] = (xb[i] - mu) * rstd;
}

// ---------------------------------------------------------------------------
extern "C" void kernel_launch(void* const* d_in, const int* in_sizes, int n_in,
                              void* d_out, int out_size, void* d_ws, size_t ws_size,
                              hipStream_t stream)
{
    (void)in_sizes; (void)n_in; (void)out_size; (void)ws_size;
    const int Nn = 4, Cc = 256, KC = 128, VC = 256, MID = 32, L = 12544;
    const size_t nMid = (size_t)Nn * MID * L;
    const size_t nK   = (size_t)Nn * KC  * L;
    const size_t nV   = (size_t)Nn * VC  * L;
    const size_t nCtx = (size_t)Nn * 8 * 16 * 32;
    const int TOT = (int)nV;

    float* ws = (float*)d_ws;
    float* mid1 = ws;
    float* mid2 = mid1 + nMid;
    float* Kb   = mid2 + nMid;
    float* Qb   = Kb + nK;
    float* Vb   = Qb + nK;          // reused as xbuf after attention
    float* ctx  = Vb + nV;
    float* agg  = ctx + nCtx;
    float* rpj  = agg + nV;
    float* part = rpj + nV;         // 1024 sums + 1024 sumsq + {mu, rstd}

    const float* input_ = (const float*)d_in[0];
    float* out = (float*)d_out;

    auto run_pw = [&](const float* X, const float* Wt, const float* Bias, float* Y,
                      int M, int K) {
        int mtiles = M / 16;
        int l4 = L / 64;                         // 196
        int blocks = Nn * mtiles * l4;
        pw_gemm_wmma<<<blocks, 128, 0, stream>>>(X, Wt, Bias, Y, M, K, L, mtiles, l4);
    };
    auto run_dw = [&](const float* X, const float* Wt, const float* Bias, float* Y) {
        int total = (int)nMid;
        dw_conv3<<<(total + 255) / 256, 256, 0, stream>>>(X, Wt, Bias, Y, total);
    };
    auto run_block = [&](int base, const float* X, float* Y, int M) {
        const float* pw1_w = (const float*)d_in[base + 0];
        const float* pw1_b = (const float*)d_in[base + 1];
        const float* dw_w  = (const float*)d_in[base + 2];
        const float* dw_b  = (const float*)d_in[base + 3];
        const float* pw2_w = (const float*)d_in[base + 4];
        const float* pw2_b = (const float*)d_in[base + 5];
        run_pw(X, pw1_w, pw1_b, mid1, MID, Cc);
        run_dw(mid1, dw_w, dw_b, mid2);
        run_pw(mid2, pw2_w, pw2_b, Y, M, MID);
    };

    run_block(1,  input_, Kb, KC);     // keys
    run_block(7,  input_, Qb, KC);     // queries
    run_block(13, input_, Vb, VC);     // values

    softmax_rows<<<Nn * KC, 256, 0, stream>>>(Kb, L);
    {
        int total = Nn * 8 * L;
        softmax_chan<<<(total + 255) / 256, 256, 0, stream>>>(Qb, total, L);
    }

    ctx_wmma<<<16, 128, 0, stream>>>(Kb, Vb, ctx, L);
    {
        int total = Nn * 8 * 2 * (L / 16);
        att_wmma<<<(total + 3) / 4, 128, 0, stream>>>(ctx, Qb, agg, L, total);
    }

    run_block(19, agg, rpj, Cc);       // reprojection

    float* xbuf = Vb;
    add_reduce<<<1024, 256, 0, stream>>>(rpj, input_, xbuf, part, TOT);
    finalize_stats<<<1, 256, 0, stream>>>(part, TOT);
    normalize_out<<<(TOT + 255) / 256, 256, 0, stream>>>(xbuf, part, out, TOT);
}
